// SelfAttentionHead_55654186222043
// MI455X (gfx1250) — matrix operand order repaired and checked by reading, not compiled, not run on hardware
//
#include <hip/hip_runtime.h>

// ---------------------------------------------------------------------------
// SelfAttentionHead for MI455X (gfx1250): bf16 WMMA pipeline, f32 accumulate,
// async global->LDS double-buffered tiles shared by 8 waves per workgroup.
//   B=4, S=4096, D=1024
//   1) convert x, Wq, Wk, Wv to bf16
//   2) qT = (x@Wq^T+bq)^T [b,d,s] bf16 ; kT likewise ; v = x@Wv^T+bv [b,s,d]
//   3) scores[b,d,e] = (1/32) * sum_s qT[b,d,s]*kT[b,e,s]  (f32)
//   4) w = softmax_rows(scores) -> bf16 [b,d,e]
//   5) out[b,s,d] = sum_e v[b,s,e]*w[b,d,e]                (f32)
// All GEMMs: C[m,n] = sum_k A[m,k]*Bm[n,k], both operands K-contiguous.
// Workgroup macro-tile 128(M)x256(N), per-wave 64x64, K-step 32.
// ---------------------------------------------------------------------------

typedef __attribute__((ext_vector_type(16))) __bf16 v16bf;
typedef __attribute__((ext_vector_type(8)))  __bf16 v8bf;
typedef __attribute__((ext_vector_type(8)))  float  v8f;
typedef __attribute__((ext_vector_type(4)))  float  v4f;

// 16B async copy: global -> LDS, tracked by ASYNCcnt (CDNA5 async path).
__device__ __forceinline__ void async_copy16(unsigned int lds_off, const void* gp) {
  asm volatile("global_load_async_to_lds_b128 %0, %1, off"
               :: "v"(lds_off), "v"((unsigned long long)(uintptr_t)gp)
               : "memory");
}
__device__ __forceinline__ void wait_async0() {
  asm volatile("s_wait_asynccnt 0x0" ::: "memory");
}

// -------------------------- f32 -> bf16 convert ----------------------------
__global__ __launch_bounds__(256) void f32_to_bf16_kernel(
    const float* __restrict__ in, __bf16* __restrict__ out, size_t n) {
  size_t i = ((size_t)blockIdx.x * blockDim.x + threadIdx.x) * 8;
  if (i >= n) return;
  v4f a = *(const v4f*)(in + i);
  v4f b = *(const v4f*)(in + i + 4);
  v8bf o;
#pragma unroll
  for (int j = 0; j < 4; ++j) {
    o[j]     = (__bf16)a[j];
    o[4 + j] = (__bf16)b[j];
  }
  *(v8bf*)(out + i) = o;
}

// ------------------------------- GEMM --------------------------------------
// MODE 0: f32  out, C[b*cs + m*ldc + n] = c*scale + bias[n]
// MODE 1: bf16 out, same addressing
// MODE 2: bf16 transposed out, C[b*cs + n*ldc + m]  (ldc = leading dim over m)
template <int MODE>
__global__ __launch_bounds__(256) void gemm_wmma_bf16(
    const __bf16* __restrict__ A, const __bf16* __restrict__ Bm,
    const float* __restrict__ bias, void* __restrict__ Cout,
    int M, int N, int K, int lda, int ldb, int ldc,
    unsigned long long aStride, unsigned long long bStride,
    unsigned long long cStride, float scale) {
  // Double-buffered LDS: per buffer A tile 128x32 bf16 (8KB) + B tile 256x32 (16KB).
  __shared__ __align__(16) unsigned char smem[2][24576];

  const int tid  = threadIdx.x;
  const int lane = tid & 31;
  const int wv   = tid >> 5;   // 0..7
  const int wm   = wv >> 2;    // 0..1 : M sub-tile
  const int wn   = wv & 3;     // 0..3 : N sub-tile
  const int lr   = lane & 15;
  const int hi   = lane >> 4;

  const int tilesN = N >> 8;                       // 256-wide N macro tiles
  const int btn = (blockIdx.x % tilesN) << 8;
  const int btm = (blockIdx.x / tilesN) << 7;      // 128-tall M macro tiles
  const int tm = btm + wm * 64;
  const int tn = btn + wn * 64;

  const int b = blockIdx.z;
  A  += (size_t)b * aStride;
  Bm += (size_t)b * bStride;

  unsigned int ldsA[2], ldsB[2];
  ldsA[0] = (unsigned int)(uintptr_t)&smem[0][0];
  ldsA[1] = (unsigned int)(uintptr_t)&smem[1][0];
  ldsB[0] = ldsA[0] + 8192;
  ldsB[1] = ldsA[1] + 8192;

  // Stage one K-step tile: A 512 x 16B chunks, B 1024 x 16B chunks,
  // row-major in LDS (row stride 64B = 32 bf16).
  auto stage = [&](int k0, int bufi) {
#pragma unroll
    for (int c = tid; c < 512; c += 256) {
      const int row = c >> 2, kc = c & 3;
      async_copy16(ldsA[bufi] + c * 16,
                   A + (size_t)(btm + row) * (size_t)lda + k0 + kc * 8);
    }
#pragma unroll
    for (int c = tid; c < 1024; c += 256) {
      const int row = c >> 2, kc = c & 3;
      async_copy16(ldsB[bufi] + c * 16,
                   Bm + (size_t)(btn + row) * (size_t)ldb + k0 + kc * 8);
    }
  };

  v8f c[4][4] = {};
  const int nk = K >> 5;

  stage(0, 0);
  wait_async0();
  __syncthreads();

  for (int t = 0; t < nk; ++t) {
    if (t + 1 < nk) stage((t + 1) << 5, (t + 1) & 1);

    const unsigned char* As = smem[t & 1];
    const unsigned char* Bs = smem[t & 1] + 8192;

    // A fragments from LDS: lane holds row; halves at K-bytes hi*16 and 32+hi*16.
    v16bf a[4];
#pragma unroll
    for (int i = 0; i < 4; ++i) {
      const __bf16* pa = (const __bf16*)(As + (wm * 64 + i * 16 + lr) * 64 + hi * 16);
      v8bf lo = *(const v8bf*)pa;
      v8bf h2 = *(const v8bf*)(pa + 16);
      a[i] = __builtin_shufflevector(lo, h2, 0, 1, 2, 3, 4, 5, 6, 7,
                                             8, 9, 10, 11, 12, 13, 14, 15);
    }
    // B fragments: lane holds column; halves at K-bytes hi*16 and 32+hi*16.
#pragma unroll
    for (int j = 0; j < 4; ++j) {
      const __bf16* pb = (const __bf16*)(Bs + (wn * 64 + j * 16 + lr) * 64 + hi * 16);
      v8bf lo = *(const v8bf*)pb;
      v8bf h2 = *(const v8bf*)(pb + 16);
      v16bf bb = __builtin_shufflevector(lo, h2, 0, 1, 2, 3, 4, 5, 6, 7,
                                                 8, 9, 10, 11, 12, 13, 14, 15);
#pragma unroll
      for (int i = 0; i < 4; ++i) {
        c[i][j] = __builtin_amdgcn_wmma_f32_16x16x32_bf16(
            false, a[i], false, bb, (short)0, c[i][j], false, false);
      }
    }

    if (t + 1 < nk) {
      wait_async0();      // my async writes into buf[(t+1)&1] have landed
      __syncthreads();    // everyone's landed + all reads of buf[t&1] done
    }
  }

  // Epilogue. C/D layout: lane holds column n; VGPR r -> m = r + hi*8.
#pragma unroll
  for (int j = 0; j < 4; ++j) {
    const int n = tn + j * 16 + lr;
    const float bv = bias ? bias[n] : 0.0f;
#pragma unroll
    for (int i = 0; i < 4; ++i) {
      if (MODE == 2) {
        __bf16* ct = (__bf16*)Cout + (size_t)b * cStride +
                     (size_t)n * (size_t)ldc + (tm + i * 16 + hi * 8);
        v8bf o;
#pragma unroll
        for (int r = 0; r < 8; ++r) o[r] = (__bf16)(c[i][j][r] * scale + bv);
        *(v8bf*)ct = o;
      } else {
#pragma unroll
        for (int r = 0; r < 8; ++r) {
          const int m = tm + i * 16 + hi * 8 + r;
          const float val = c[i][j][r] * scale + bv;
          const size_t idx = (size_t)b * cStride + (size_t)m * (size_t)ldc + n;
          if (MODE == 0) ((float*)Cout)[idx] = val;
          else           ((__bf16*)Cout)[idx] = (__bf16)val;
        }
      }
    }
  }
}

// ----------------------------- row softmax ---------------------------------
__global__ __launch_bounds__(256) void softmax_rows_kernel(
    const float* __restrict__ in, __bf16* __restrict__ out, int cols) {
  const int row = blockIdx.x;
  const float* r = in + (size_t)row * cols;
  __bf16* o = out + (size_t)row * cols;
  __shared__ float red[256];
  const int tid = threadIdx.x;

  float vmax = -3.402823466e38f;
  for (int c = tid; c < cols; c += 256) vmax = fmaxf(vmax, r[c]);
  red[tid] = vmax;
  __syncthreads();
  for (int s = 128; s > 0; s >>= 1) {
    if (tid < s) red[tid] = fmaxf(red[tid], red[tid + s]);
    __syncthreads();
  }
  vmax = red[0];
  __syncthreads();

  float e[4];
  float sum = 0.0f;
#pragma unroll
  for (int i = 0; i < 4; ++i) {
    e[i] = __expf(r[tid + i * 256] - vmax);
    sum += e[i];
  }
  red[tid] = sum;
  __syncthreads();
  for (int s = 128; s > 0; s >>= 1) {
    if (tid < s) red[tid] += red[tid + s];
    __syncthreads();
  }
  const float inv = 1.0f / red[0];
#pragma unroll
  for (int i = 0; i < 4; ++i) o[tid + i * 256] = (__bf16)(e[i] * inv);
}

// ------------------------------ launcher -----------------------------------
extern "C" void kernel_launch(void* const* d_in, const int* in_sizes, int n_in,
                              void* d_out, int out_size, void* d_ws, size_t ws_size,
                              hipStream_t stream) {
  (void)in_sizes; (void)n_in; (void)out_size; (void)ws_size;
  constexpr int Bn = 4, S = 4096, D = 1024;
  constexpr size_t MS = (size_t)Bn * S * D;  // 16,777,216 elements

  const float* x  = (const float*)d_in[0];
  const float* Wk = (const float*)d_in[1];
  const float* bk = (const float*)d_in[2];
  const float* Wq = (const float*)d_in[3];
  const float* bq = (const float*)d_in[4];
  const float* Wv = (const float*)d_in[5];
  const float* bv = (const float*)d_in[6];
  float* out = (float*)d_out;

  char* ws = (char*)d_ws;
  size_t off = 0;
  __bf16* xb  = (__bf16*)(ws + off); off += MS * 2;                 // 32 MiB
  __bf16* Wqb = (__bf16*)(ws + off); off += (size_t)D * D * 2;      //  2 MiB
  __bf16* Wkb = (__bf16*)(ws + off); off += (size_t)D * D * 2;
  __bf16* Wvb = (__bf16*)(ws + off); off += (size_t)D * D * 2;
  __bf16* qT  = (__bf16*)(ws + off); off += MS * 2;                 // 32 MiB
  __bf16* kT  = (__bf16*)(ws + off); off += MS * 2;                 // 32 MiB
  __bf16* vb  = (__bf16*)(ws + off); off += MS * 2;                 // 32 MiB
  // After the projections xb is dead: alias scores (16 MiB f32) + w (8 MiB bf16).
  float*  scoresF = (float*)xb;
  __bf16* wb = (__bf16*)((char*)xb + (size_t)Bn * D * D * 4);

  // 1) convert to bf16
  f32_to_bf16_kernel<<<(unsigned)(MS / 2048), 256, 0, stream>>>(x, xb, MS);
  f32_to_bf16_kernel<<<(unsigned)((size_t)D * D / 2048), 256, 0, stream>>>(Wq, Wqb, (size_t)D * D);
  f32_to_bf16_kernel<<<(unsigned)((size_t)D * D / 2048), 256, 0, stream>>>(Wk, Wkb, (size_t)D * D);
  f32_to_bf16_kernel<<<(unsigned)((size_t)D * D / 2048), 256, 0, stream>>>(Wv, Wvb, (size_t)D * D);

  // 2) projections: per batch M=S, N=D, K=D. Macro tiles: (4096/128)*(1024/256)=128 blocks.
  const dim3 gProj(128, 1, Bn);
  gemm_wmma_bf16<2><<<gProj, 256, 0, stream>>>(
      xb, Wqb, bq, qT, S, D, D, D, D, /*ldc over m*/ S,
      (unsigned long long)S * D, 0ULL, (unsigned long long)D * S, 1.0f);
  gemm_wmma_bf16<2><<<gProj, 256, 0, stream>>>(
      xb, Wkb, bk, kT, S, D, D, D, D, S,
      (unsigned long long)S * D, 0ULL, (unsigned long long)D * S, 1.0f);
  gemm_wmma_bf16<1><<<gProj, 256, 0, stream>>>(
      xb, Wvb, bv, vb, S, D, D, D, D, D,
      (unsigned long long)S * D, 0ULL, (unsigned long long)S * D, 1.0f);

  // 3) scores: per batch M=N=D, K=S. (1024/128)*(1024/256)=32 blocks.
  const dim3 gScore(32, 1, Bn);
  gemm_wmma_bf16<0><<<gScore, 256, 0, stream>>>(
      qT, kT, nullptr, scoresF, D, D, S, S, S, D,
      (unsigned long long)D * S, (unsigned long long)D * S,
      (unsigned long long)D * D, 0.03125f);

  // 4) softmax over the feature axis e (rows of [B*D, D])
  softmax_rows_kernel<<<Bn * D, 256, 0, stream>>>(scoresF, wb, D);

  // 5) out[b,s,d] = sum_e v[s,e]*w[d,e]: per batch M=S, N=D, K=D.
  gemm_wmma_bf16<0><<<gProj, 256, 0, stream>>>(
      vb, wb, nullptr, out, S, D, D, D, D, D,
      (unsigned long long)S * D, (unsigned long long)D * D,
      (unsigned long long)S * D, 1.0f);
}
